// RNN_63436666962428
// MI455X (gfx1250) — compile-verified, matrix operand order checked by reading, not run on hardware
//
#include <hip/hip_runtime.h>
#include <hip/hip_bf16.h>
#include <math.h>

// ---- problem constants (match reference) ----
constexpr int Nn = 64;
constexpr int Tt = 512;
constexpr int Dd = 1024;
constexpr int Hh = 1024;
constexpr int NWG_SCAN = Hh / 64;   // 16 persistent workgroups in the scan

typedef __bf16 v16bf __attribute__((ext_vector_type(16)));
typedef __bf16 v8bf  __attribute__((ext_vector_type(8)));
typedef __bf16 v4bf  __attribute__((ext_vector_type(4)));
typedef float  v8f   __attribute__((ext_vector_type(8)));

// gfx1250 has a hardware v_tanh_f32 (TRANS op); use it on the scan's serial
// critical path. __has_builtin guard keeps the compile safe either way.
__device__ __forceinline__ float fast_tanh(float x) {
#if __has_builtin(__builtin_amdgcn_tanhf)
  return __builtin_amdgcn_tanhf(x);
#else
  return tanhf(x);
#endif
}

// ---------------------------------------------------------------------------
// f32 -> bf16 bulk convert (vectorized: float4 in, 4x bf16 out)
// ---------------------------------------------------------------------------
__global__ void __launch_bounds__(256) k_cvt_bf16(const float* __restrict__ in,
                                                  __bf16* __restrict__ out,
                                                  size_t n4) {
  size_t i = ((size_t)blockIdx.x * blockDim.x + threadIdx.x);
  if (i < n4) {
    float4 v = reinterpret_cast<const float4*>(in)[i];
    v4bf o;
    o[0] = (__bf16)v.x; o[1] = (__bf16)v.y; o[2] = (__bf16)v.z; o[3] = (__bf16)v.w;
    reinterpret_cast<v4bf*>(out)[i] = o;
  }
}

// ---------------------------------------------------------------------------
// LDS-tiled transpose + f32->bf16 convert: out[c][r] = (bf16)in[r][c]
// block (32,8), grid (cols/32, rows/32)
// ---------------------------------------------------------------------------
__global__ void __launch_bounds__(256) k_transpose_cvt(const float* __restrict__ in,
                                                       __bf16* __restrict__ out,
                                                       int rows, int cols) {
  __shared__ float tile[32][33];   // +1 pad: no LDS bank conflicts
  const int bx = blockIdx.x * 32;  // col base
  const int by = blockIdx.y * 32;  // row base
  const int tx = threadIdx.x, ty = threadIdx.y;
  #pragma unroll
  for (int i = ty; i < 32; i += 8)
    tile[i][tx] = in[(size_t)(by + i) * cols + (bx + tx)];
  __syncthreads();
  #pragma unroll
  for (int i = ty; i < 32; i += 8)
    out[(size_t)(bx + i) * rows + (by + tx)] = (__bf16)tile[tx][i];
}

// ---------------------------------------------------------------------------
// Load a 16x32 bf16 A/B fragment (row-major, leading dim ld) per ISA layout:
//   lanes 0-15  (row = lane)    : elems[0..7]=K 0..7,  elems[8..15]=K 16..23
//   lanes 16-31 (row = lane-16) : elems[0..7]=K 8..15, elems[8..15]=K 24..31
// Two global_load_b128 per lane.
// ---------------------------------------------------------------------------
__device__ __forceinline__ v16bf load_frag16x32(const __bf16* __restrict__ base,
                                                int ld, int k0) {
  const int lane = threadIdx.x & 31;
  const int r = lane & 15;
  const int khalf = (lane & 16) ? 8 : 0;
  const __bf16* p = base + (size_t)r * ld + k0 + khalf;
  v16bf f;
  reinterpret_cast<v8bf*>(&f)[0] = *reinterpret_cast<const v8bf*>(p);
  reinterpret_cast<v8bf*>(&f)[1] = *reinterpret_cast<const v8bf*>(p + 16);
  return f;
}

#define WMMA_BF16(A, B, C) \
  __builtin_amdgcn_wmma_f32_16x16x32_bf16(false, (A), false, (B), (short)0, (C), false, false)

// ---------------------------------------------------------------------------
// Phase 1: xW = x @ Wx + b  (M = N*T = 32768, K = D, N = H), f32 out -> d_out.
// 256 threads = 8 waves; WG tile 64x256; each wave 32x64 (2x4 WMMA tiles):
// 12 b128 loads per 8 WMMAs (1.5 loads/WMMA vs 2.0 with 2x2 blocking).
// ---------------------------------------------------------------------------
__global__ void __launch_bounds__(256) k_gemm_xw(const __bf16* __restrict__ A,
                                                 const __bf16* __restrict__ BT,
                                                 const float* __restrict__ bias,
                                                 float* __restrict__ C) {
  const int w  = threadIdx.x >> 5;
  const int wr = w & 1;        // 0..1 : 32-row block
  const int wc = w >> 1;       // 0..3 : 64-col block
  const int r0 = blockIdx.x * 64  + wr * 32;
  const int c0 = blockIdx.y * 256 + wc * 64;

  const __bf16* Ar[2];
  #pragma unroll
  for (int i = 0; i < 2; ++i) Ar[i] = A + (size_t)(r0 + i * 16) * Dd;
  const __bf16* Bc[4];
  #pragma unroll
  for (int j = 0; j < 4; ++j) Bc[j] = BT + (size_t)(c0 + j * 16) * Dd;

  v8f acc[2][4] = {};
  for (int k = 0; k < Dd; k += 32) {
    v16bf a[2], b[4];
    #pragma unroll
    for (int i = 0; i < 2; ++i) a[i] = load_frag16x32(Ar[i], Dd, k);
    #pragma unroll
    for (int j = 0; j < 4; ++j) b[j] = load_frag16x32(Bc[j], Dd, k);
    #pragma unroll
    for (int i = 0; i < 2; ++i)
      #pragma unroll
      for (int j = 0; j < 4; ++j)
        acc[i][j] = WMMA_BF16(a[i], b[j], acc[i][j]);
  }

  const int lane = threadIdx.x & 31;
  const int cl = lane & 15;
  const int rl = (lane & 16) ? 8 : 0;
  #pragma unroll
  for (int it = 0; it < 2; ++it)
    #pragma unroll
    for (int jt = 0; jt < 4; ++jt) {
      const int col = c0 + jt * 16 + cl;
      const int row = r0 + it * 16 + rl;
      const float bv = bias[col];
      #pragma unroll
      for (int i = 0; i < 8; ++i)
        C[(size_t)(row + i) * Hh + col] = acc[it][jt][i] + bv;
    }
}

// ---------------------------------------------------------------------------
// Phase 2: persistent recurrent scan. 16 WGs x 128 threads (4 waves).
// WG owns a 64-col slab of H; wave does 32x32 (2x2 WMMA tiles) of the
// 64x64 (batch x slab) step-GEMM h_prev @ WhT_slab. Per step:
//   prefetch xW_{t+1} slice   (global_prefetch_b8, hides HBM latency under K loop)
//   acc = h_prev(bf16) @ Wh(bf16)   [f32 accum, 32 K-chunks x 4 wmma]
//   v   = tanh(acc + xW_t)          [v_tanh_f32; xW_t read from d_out, f32]
//   d_out slice <- v (f32); h_next <- bf16(v)
// then a device-wide monotonic atomic barrier (count >= 16*(t+1)).
// Wh (2MB) + h buffers (2x128KB) stay L2-resident.
// ---------------------------------------------------------------------------
__global__ void __launch_bounds__(128) k_scan(__bf16* __restrict__ hb0,
                                              __bf16* __restrict__ hb1,
                                              const __bf16* __restrict__ WhT,
                                              float* out,
                                              unsigned* __restrict__ cnt) {
  const int w  = threadIdx.x >> 5;  // 0..3
  const int wr = w & 1;
  const int wc = w >> 1;
  const int r0 = wr * 32;                        // batch rows (0..63)
  const int c0 = blockIdx.x * 64 + wc * 32;      // H columns
  const int lane = threadIdx.x & 31;
  const int cl = lane & 15;
  const int rl = (lane & 16) ? 8 : 0;

  const __bf16* Bc0 = WhT + (size_t)c0 * Hh;
  const __bf16* Bc1 = WhT + (size_t)(c0 + 16) * Hh;
  __bf16* bufs[2] = { hb0, hb1 };

  for (int t = 0; t < Tt; ++t) {
    const __bf16* hp = bufs[t & 1];
    __bf16* hn = bufs[(t + 1) & 1];
    const __bf16* Ar0 = hp + (size_t)r0 * Hh;
    const __bf16* Ar1 = hp + (size_t)(r0 + 16) * Hh;

    // Prefetch next timestep's xW slice so the epilogue reads hit L2.
    // (16 consecutive lanes share a 64B line; HW dedups.)
    if (t + 1 < Tt) {
      #pragma unroll
      for (int it = 0; it < 2; ++it)
        #pragma unroll
        for (int jt = 0; jt < 2; ++jt) {
          const int col = c0 + jt * 16 + cl;
          const int row = r0 + it * 16 + rl;
          #pragma unroll
          for (int i = 0; i < 8; ++i)
            __builtin_prefetch(&out[((size_t)(row + i) * Tt + (t + 1)) * Hh + col], 0, 1);
        }
    }

    v8f acc[2][2] = {};
    for (int k = 0; k < Hh; k += 32) {
      v16bf a0 = load_frag16x32(Ar0, Hh, k);
      v16bf a1 = load_frag16x32(Ar1, Hh, k);
      v16bf b0 = load_frag16x32(Bc0, Hh, k);
      v16bf b1 = load_frag16x32(Bc1, Hh, k);
      acc[0][0] = WMMA_BF16(a0, b0, acc[0][0]);
      acc[0][1] = WMMA_BF16(a0, b1, acc[0][1]);
      acc[1][0] = WMMA_BF16(a1, b0, acc[1][0]);
      acc[1][1] = WMMA_BF16(a1, b1, acc[1][1]);
    }

    #pragma unroll
    for (int it = 0; it < 2; ++it)
      #pragma unroll
      for (int jt = 0; jt < 2; ++jt) {
        const int col = c0 + jt * 16 + cl;
        const int row = r0 + it * 16 + rl;
        #pragma unroll
        for (int i = 0; i < 8; ++i) {
          const int n = row + i;
          const size_t oidx = ((size_t)n * Tt + t) * Hh + col;
          const float v = fast_tanh(acc[it][jt][i] + out[oidx]);  // xW_t read, then overwritten
          out[oidx] = v;
          hn[(size_t)n * Hh + col] = (__bf16)v;
        }
      }

    // device-wide barrier across the 16 persistent WGs (monotonic counter)
    __syncthreads();
    if (threadIdx.x == 0) {
      __threadfence();                       // release h_next stores
      atomicAdd(cnt, 1u);
      const unsigned target = (unsigned)NWG_SCAN * (unsigned)(t + 1);
      while (__hip_atomic_load(cnt, __ATOMIC_ACQUIRE, __HIP_MEMORY_SCOPE_AGENT) < target)
        __builtin_amdgcn_s_sleep(1);
    }
    __syncthreads();
  }
}

// ---------------------------------------------------------------------------
// Launcher. Workspace layout (bytes):
//   [0)            x_bf16   : 32768*1024*2 = 64 MiB
//   [64Mi)         WxT_bf16 : 2 MiB  (column-major: [h][d])
//   [66Mi)         WhT_bf16 : 2 MiB  (column-major: [h_out][h_in])
//   [68Mi)         hbuf0    : 128 KiB (bf16 hidden state, buffer A)
//   [68Mi+128Ki)   hbuf1    : 128 KiB (buffer B)
//   [68Mi+256Ki)   barrier  : 64 B
// ---------------------------------------------------------------------------
extern "C" void kernel_launch(void* const* d_in, const int* in_sizes, int n_in,
                              void* d_out, int out_size, void* d_ws, size_t ws_size,
                              hipStream_t stream) {
  (void)in_sizes; (void)n_in; (void)out_size; (void)ws_size;

  const float* x  = (const float*)d_in[0];   // (N,T,D)
  const float* h0 = (const float*)d_in[1];   // (N,H)
  const float* Wx = (const float*)d_in[2];   // (D,H)
  const float* Wh = (const float*)d_in[3];   // (H,H)
  const float* b  = (const float*)d_in[4];   // (H,)
  float* out = (float*)d_out;                // (N,T,H)

  char* ws = (char*)d_ws;
  const size_t OFF_XBF = 0;
  const size_t OFF_WXT = OFF_XBF + (size_t)Nn * Tt * Dd * 2;   // 64 MiB
  const size_t OFF_WHT = OFF_WXT + (size_t)Dd * Hh * 2;        // +2 MiB
  const size_t OFF_HB0 = OFF_WHT + (size_t)Hh * Hh * 2;        // +2 MiB
  const size_t OFF_HB1 = OFF_HB0 + (size_t)Nn * Hh * 2;        // +128 KiB
  const size_t OFF_CNT = OFF_HB1 + (size_t)Nn * Hh * 2;        // +128 KiB

  __bf16* xbf = (__bf16*)(ws + OFF_XBF);
  __bf16* WxT = (__bf16*)(ws + OFF_WXT);
  __bf16* WhT = (__bf16*)(ws + OFF_WHT);
  __bf16* hb0 = (__bf16*)(ws + OFF_HB0);
  __bf16* hb1 = (__bf16*)(ws + OFF_HB1);
  unsigned* cnt = (unsigned*)(ws + OFF_CNT);

  hipMemsetAsync(cnt, 0, 64, stream);  // reset barrier each call (graph-safe)

  // bf16 conversions
  {
    const size_t n4 = (size_t)Nn * Tt * Dd / 4;          // 8M float4s
    k_cvt_bf16<<<(unsigned)((n4 + 255) / 256), 256, 0, stream>>>(x, xbf, n4);
  }
  {
    const size_t n4 = (size_t)Nn * Hh / 4;
    k_cvt_bf16<<<(unsigned)((n4 + 255) / 256), 256, 0, stream>>>(h0, hb0, n4);
  }
  dim3 tpb(32, 8);
  k_transpose_cvt<<<dim3(Hh / 32, Dd / 32), tpb, 0, stream>>>(Wx, WxT, Dd, Hh);
  k_transpose_cvt<<<dim3(Hh / 32, Hh / 32), tpb, 0, stream>>>(Wh, WhT, Hh, Hh);

  // Phase 1: xW + b -> d_out  (M=32768 rows in blocks of 64, H in blocks of 256)
  k_gemm_xw<<<dim3((Nn * Tt) / 64, Hh / 256), 256, 0, stream>>>(xbf, WxT, b, out);

  // Phase 2: persistent recurrent scan
  k_scan<<<NWG_SCAN, 128, 0, stream>>>(hb0, hb1, WhT, out, cnt);
}
